// SoluModel_3049426780886
// MI455X (gfx1250) — matrix operand order repaired and checked by reading, not compiled, not run on hardware
//
#include <hip/hip_runtime.h>
#include <hip/hip_bf16.h>

// ---------------------------------------------------------------------------
// SoluModel forward for MI455X (gfx1250, wave32, WMMA 16x16x32 bf16)
// B=16, L=1024, D=640, D2=1280
// ---------------------------------------------------------------------------

typedef __bf16 bf16_t;
typedef __attribute__((ext_vector_type(16))) __bf16 v16bf;
typedef __attribute__((ext_vector_type(8)))  __bf16 v8bf;
typedef __attribute__((ext_vector_type(4)))  __bf16 v4bf;
typedef __attribute__((ext_vector_type(8)))  float  v8f;

union Frag { v16bf v; v8bf h[2]; };

__device__ __forceinline__ float leaky01(float x) { return x >= 0.f ? x : 0.01f * x; }

__device__ __forceinline__ v8f wmma_bf16(v16bf a, v16bf b, v8f c) {
    return __builtin_amdgcn_wmma_f32_16x16x32_bf16(false, a, false, b, (short)0, c, false, false);
}

// ---------------------------------------------------------------------------
// Weight / input packing
// ---------------------------------------------------------------------------
__global__ void pack_conv_w(const float* __restrict__ w, bf16_t* __restrict__ wp,
                            int O, int I) {
    // w [O,I,3] f32  ->  wp [3,O,I] bf16
    long n = (long)O * I * 3;
    for (long idx = blockIdx.x * (long)blockDim.x + threadIdx.x; idx < n;
         idx += (long)gridDim.x * blockDim.x) {
        int tap = (int)(idx % 3);
        int i   = (int)((idx / 3) % I);
        int o   = (int)(idx / (3L * I));
        wp[((long)tap * O + o) * I + i] = (bf16_t)w[idx];
    }
}

__global__ void pack_mlp(const float* __restrict__ w, bf16_t* __restrict__ wT,
                         int O, int K) {
    // w [O,K] -> wT [K,O] bf16
    long n = (long)O * K;
    for (long idx = blockIdx.x * (long)blockDim.x + threadIdx.x; idx < n;
         idx += (long)gridDim.x * blockDim.x) {
        int k = (int)(idx % K);
        int o = (int)(idx / K);
        wT[(long)k * O + o] = (bf16_t)w[idx];
    }
}

__global__ void prep_feats(const float* __restrict__ f, bf16_t* __restrict__ fT,
                           bf16_t* __restrict__ fb, int Bn, int L, int D) {
    // f [B,L,D] f32 -> fb [B,L,D] bf16, fT [B,D,L] bf16
    long total = (long)Bn * L * D;
    for (long i = blockIdx.x * (long)blockDim.x + threadIdx.x; i < total;
         i += (long)gridDim.x * blockDim.x) {
        int d = (int)(i % D);
        long t = i / D;
        int l = (int)(t % L);
        int b = (int)(t / L);
        bf16_t v = (bf16_t)f[i];
        fb[i] = v;
        fT[((long)b * D + d) * L + l] = v;
    }
}

// ---------------------------------------------------------------------------
// Conv1d (k=3, SAME) as 3-tap WMMA GEMM.  x [B,Cin,L] bf16, wp [3,Cout,Cin],
// y [B,Cout,L] f32.  Block = 256 = 8 waves; WG tile = 128(O) x 32(L).
// Each wave: 16(O) x 32(L), two accumulators, A reused across both.
// LDS x-tile transposed [pos][chan] so B-fragments are 2x ds_load_b128.
// ---------------------------------------------------------------------------
__global__ __launch_bounds__(256) void conv_gemm(
    const bf16_t* __restrict__ x, const bf16_t* __restrict__ wp,
    const float* __restrict__ bias, float* __restrict__ y,
    int Cin, int Cout, int L) {
    const int l0   = blockIdx.x * 32;
    const int oblk = blockIdx.y * 128;
    const int b    = blockIdx.z;
    const int tid  = threadIdx.x;
    const int wave = tid >> 5;
    const int lane = tid & 31;
    const int n    = lane & 15;
    const int half = lane >> 4;
    const int obase = oblk + wave * 16;

    // [column (position, 34 incl halo)][K (channel, 32)], 64B rows
    __shared__ bf16_t lxT[34][32];

    const bf16_t* xb = x + (long)b * Cin * L;
    v8f acc0 = {}, acc1 = {};

    for (int k0 = 0; k0 < Cin; k0 += 32) {
        __syncthreads();
        {   // interior: 32 chan x 32 positions, v4bf global loads (coalesced)
            int r  = tid >> 3;            // channel row 0..31
            int c4 = 1 + (tid & 7) * 4;   // LDS col 1,5,...,29
            v4bf q = *(const v4bf*)(xb + (long)(k0 + r) * L + (l0 + c4 - 1));
            lxT[c4 + 0][r] = q[0];
            lxT[c4 + 1][r] = q[1];
            lxT[c4 + 2][r] = q[2];
            lxT[c4 + 3][r] = q[3];
        }
        if (tid < 64) {
            // halo columns 0 and 33
            int r = tid & 31;
            int c = (tid >> 5) ? 33 : 0;
            int gl = l0 + c - 1;
            bf16_t v = (bf16_t)0.0f;
            if (gl >= 0 && gl < L) v = xb[(long)(k0 + r) * L + gl];
            lxT[c][r] = v;
        }
        if (k0 + 32 < Cin && tid < 32)
            __builtin_prefetch(xb + (long)(k0 + 32 + tid) * L + l0, 0, 3);
        __syncthreads();

#pragma unroll
        for (int tap = 0; tap < 3; ++tap) {
            // A fragment (reused for both l-subtiles):
            // row m = n, K runs [half*16, +16)
            Frag au;
            const bf16_t* wrow = wp + ((long)tap * Cout + (obase + n)) * Cin + k0;
            au.h[0] = *(const v8bf*)(wrow + half * 16);
            au.h[1] = *(const v8bf*)(wrow + half * 16 + 8);
            // B fragments: element e -> K = e + 16*half, columns n+tap, n+16+tap
            Frag b0, b1;
            const bf16_t* c0p = &lxT[n + tap][0];
            const bf16_t* c1p = &lxT[n + 16 + tap][0];
            b0.h[0] = *(const v8bf*)(c0p + half * 16);
            b0.h[1] = *(const v8bf*)(c0p + half * 16 + 8);
            b1.h[0] = *(const v8bf*)(c1p + half * 16);
            b1.h[1] = *(const v8bf*)(c1p + half * 16 + 8);
            acc0 = wmma_bf16(au.v, b0.v, acc0);
            acc1 = wmma_bf16(au.v, b1.v, acc1);
        }
    }
#pragma unroll
    for (int r = 0; r < 8; ++r) {
        int m = r + half * 8;
        int o = obase + m;
        float bv = bias[o];
        y[((long)b * Cout + o) * L + l0 + n]      = acc0[r] + bv;
        y[((long)b * Cout + o) * L + l0 + 16 + n] = acc1[r] + bv;
    }
}

// Tiny conv (Cout=2) — scalar
__global__ void conv_small(const bf16_t* __restrict__ x, const float* __restrict__ w,
                           const float* __restrict__ bias, float* __restrict__ y,
                           int Cin, int L) {
    int l = blockIdx.x * blockDim.x + threadIdx.x;
    int c = blockIdx.y;
    int b = blockIdx.z;
    if (l >= L) return;
    const bf16_t* xb = x + (long)b * Cin * L;
    float acc = bias[c];
    for (int i = 0; i < Cin; ++i) {
        const float* wr = w + ((long)c * Cin + i) * 3;
        float xm = (l > 0)     ? (float)xb[(long)i * L + l - 1] : 0.f;
        float x0 =               (float)xb[(long)i * L + l];
        float xp = (l < L - 1) ? (float)xb[(long)i * L + l + 1] : 0.f;
        acc += wr[0] * xm + wr[1] * x0 + wr[2] * xp;
    }
    y[((long)b * 2 + c) * L + l] = acc;
}

// ---------------------------------------------------------------------------
// LayerNorm over (C,L) per batch: deterministic two-pass
// ---------------------------------------------------------------------------
__global__ __launch_bounds__(256) void ln_stats(const float* __restrict__ y,
                                                float* __restrict__ mr, int N) {
    const int b = blockIdx.x;
    const float* p = y + (long)b * N;
    float s = 0.f, s2 = 0.f;
    for (int i = threadIdx.x; i < N; i += 256) { float v = p[i]; s += v; s2 += v * v; }
    __shared__ float ls[256], ls2[256];
    ls[threadIdx.x] = s; ls2[threadIdx.x] = s2;
    __syncthreads();
    for (int off = 128; off > 0; off >>= 1) {
        if (threadIdx.x < off) {
            ls[threadIdx.x]  += ls[threadIdx.x + off];
            ls2[threadIdx.x] += ls2[threadIdx.x + off];
        }
        __syncthreads();
    }
    if (threadIdx.x == 0) {
        float mean = ls[0] / N;
        float var  = ls2[0] / N - mean * mean;
        mr[b * 2 + 0] = mean;
        mr[b * 2 + 1] = rsqrtf(var + 1e-5f);
    }
}

__global__ void ln_apply(const float* __restrict__ y, const float* __restrict__ mr,
                         bf16_t* __restrict__ ob, float* __restrict__ of, int N) {
    int b = blockIdx.y;
    float mean = mr[b * 2], rstd = mr[b * 2 + 1];
    const float* p = y + (long)b * N;
    for (int i = blockIdx.x * blockDim.x + threadIdx.x; i < N;
         i += gridDim.x * blockDim.x) {
        float v = leaky01((p[i] - mean) * rstd);
        if (ob) ob[(long)b * N + i] = (bf16_t)v;
        if (of) of[(long)b * N + i] = v;
    }
}

// ---------------------------------------------------------------------------
// Softmax over L for sw channels; derive invden (c=0) and w_soft (c=1)
// ---------------------------------------------------------------------------
__global__ __launch_bounds__(256) void softmax_sw(const float* __restrict__ sw,
                                                  float* __restrict__ invden,
                                                  float* __restrict__ wsoft, int L) {
    int c = blockIdx.x;
    int b = blockIdx.y;
    const float* p = sw + ((long)b * 2 + c) * L;
    __shared__ float red[256];
    float mx = -1e30f;
    for (int i = threadIdx.x; i < L; i += 256) mx = fmaxf(mx, p[i]);
    red[threadIdx.x] = mx; __syncthreads();
    for (int off = 128; off > 0; off >>= 1) {
        if (threadIdx.x < off) red[threadIdx.x] = fmaxf(red[threadIdx.x], red[threadIdx.x + off]);
        __syncthreads();
    }
    mx = red[0]; __syncthreads();
    float s = 0.f;
    for (int i = threadIdx.x; i < L; i += 256) s += __expf(p[i] - mx);
    red[threadIdx.x] = s; __syncthreads();
    for (int off = 128; off > 0; off >>= 1) {
        if (threadIdx.x < off) red[threadIdx.x] += red[threadIdx.x + off];
        __syncthreads();
    }
    float inv = 1.f / red[0];
    for (int i = threadIdx.x; i < L; i += 256) {
        float prob = __expf(p[i] - mx) * inv;
        if (c == 0) {
            float sd = 0.1f * (float)L * prob;
            invden[(long)b * L + i] = 1.f / (1e-5f + 2.f * sd * sd);
        } else {
            wsoft[(long)b * L + i] = prob;
        }
    }
}

// ---------------------------------------------------------------------------
// Gaussian context pooling: out_sa[b,i,d] = sum_j exp(-(i-j)^2*invden_i)*w[j]*f[b,j,d]
// Block = 128 = 4 waves; WG tile 16(i) x 128(d); wave tile 16 x 32 with two
// accumulators so the exp-generated A fragment is computed ONCE per 2 WMMAs.
// ---------------------------------------------------------------------------
__global__ __launch_bounds__(128) void pool_gemm(
    const bf16_t* __restrict__ fb, const float* __restrict__ invden,
    const float* __restrict__ wsoft, float* __restrict__ outsa, int L, int D) {
    const int dblk = blockIdx.x * 128;
    const int i0   = blockIdx.y * 16;
    const int b    = blockIdx.z;
    const int tid  = threadIdx.x;
    const int wave = tid >> 5;
    const int lane = tid & 31;
    const int n    = lane & 15;
    const int half = lane >> 4;

    __shared__ bf16_t lBT[128][32];  // [d-col][j-row], 64B rows
    __shared__ float  lw[32];

    const int   mi  = i0 + n;
    const float idn = invden[(long)b * L + mi];
    const bf16_t* fbb = fb + (long)b * L * D;
    const int dw = dblk + wave * 32;

    v8f acc0 = {}, acc1 = {};
    for (int j0 = 0; j0 < L; j0 += 32) {
        __syncthreads();
        for (int idx = tid; idx < 1024; idx += 128) {
            int k  = idx >> 5;            // j row 0..31
            int c4 = (idx & 31) * 4;      // d col 0,4,...,124
            v4bf q = *(const v4bf*)(fbb + (long)(j0 + k) * D + dblk + c4);
            lBT[c4 + 0][k] = q[0];
            lBT[c4 + 1][k] = q[1];
            lBT[c4 + 2][k] = q[2];
            lBT[c4 + 3][k] = q[3];
        }
        if (tid < 32) lw[tid] = wsoft[(long)b * L + j0 + tid];
        __syncthreads();

        Frag au;
#pragma unroll
        for (int e = 0; e < 16; ++e) {
            int k = e + half * 16;
            float dj = (float)(mi - (j0 + k));
            au.v[e] = (bf16_t)(__expf(-dj * dj * idn) * lw[k]);
        }
        Frag b0, b1;
        const bf16_t* c0p = &lBT[(wave & 3) * 32 + n][0];
        const bf16_t* c1p = &lBT[(wave & 3) * 32 + 16 + n][0];
        b0.h[0] = *(const v8bf*)(c0p + half * 16);
        b0.h[1] = *(const v8bf*)(c0p + half * 16 + 8);
        b1.h[0] = *(const v8bf*)(c1p + half * 16);
        b1.h[1] = *(const v8bf*)(c1p + half * 16 + 8);
        acc0 = wmma_bf16(au.v, b0.v, acc0);
        acc1 = wmma_bf16(au.v, b1.v, acc1);
    }
#pragma unroll
    for (int r = 0; r < 8; ++r) {
        int m = r + half * 8;
        outsa[((long)b * L + i0 + m) * D + dw + n]      = acc0[r];
        outsa[((long)b * L + i0 + m) * D + dw + 16 + n] = acc1[r];
    }
}

// ---------------------------------------------------------------------------
// Build concat input: cat[b,l,0:D] = out_sa + feats ; cat[b,l,D:2D] = conv^T + feats
// ---------------------------------------------------------------------------
__global__ void build_cat(const float* __restrict__ outsa, const bf16_t* __restrict__ conv,
                          const float* __restrict__ feats, bf16_t* __restrict__ cat,
                          int L, int D) {
    int l = blockIdx.x, b = blockIdx.y;
    const float* fr = feats + ((long)b * L + l) * D;
    const float* sr = outsa + ((long)b * L + l) * D;
    bf16_t* cr = cat + ((long)b * L + l) * 2 * D;
    for (int d = threadIdx.x; d < D; d += blockDim.x) {
        cr[d] = (bf16_t)(sr[d] + fr[d]);
        float cv = (float)conv[((long)b * D + d) * L + l];
        cr[D + d] = (bf16_t)(cv + fr[d]);
    }
}

// ---------------------------------------------------------------------------
// MLP: out[m,o] = leaky(cat[m,:] . wT[:,o] + b[o]).  Block = 128 = 4 waves;
// WG tile 16(m) x 128(o); wave tile 16 x 32, A reused across two accumulators.
// ---------------------------------------------------------------------------
__global__ __launch_bounds__(128) void mlp_gemm(
    const bf16_t* __restrict__ cat, const bf16_t* __restrict__ wT,
    const float* __restrict__ bias, float* __restrict__ out,
    int M, int K, int N) {
    const int nblk = blockIdx.x * 128;
    const int m0   = blockIdx.y * 16;
    const int tid  = threadIdx.x;
    const int wave = tid >> 5;
    const int lane = tid & 31;
    const int n    = lane & 15;
    const int half = lane >> 4;

    __shared__ bf16_t lBT[128][32];  // [o-col][k-row]
    __shared__ bf16_t lA[16][40];    // row stride 80B (16B multiple)

    v8f acc0 = {}, acc1 = {};
    for (int k0 = 0; k0 < K; k0 += 32) {
        __syncthreads();
        for (int idx = tid; idx < 1024; idx += 128) {
            int r  = idx >> 5;            // k row 0..31
            int c4 = (idx & 31) * 4;      // o col
            v4bf q = *(const v4bf*)(wT + (long)(k0 + r) * N + nblk + c4);
            lBT[c4 + 0][r] = q[0];
            lBT[c4 + 1][r] = q[1];
            lBT[c4 + 2][r] = q[2];
            lBT[c4 + 3][r] = q[3];
        }
        {
            int r  = tid >> 3;            // m row 0..15
            int c4 = (tid & 7) * 4;       // k col
            v4bf q = *(const v4bf*)(cat + (long)(m0 + r) * K + k0 + c4);
            *(v4bf*)&lA[r][c4] = q;
        }
        __syncthreads();
        Frag au;
        const bf16_t* arow = &lA[n][0];
        au.h[0] = *(const v8bf*)(arow + half * 16);
        au.h[1] = *(const v8bf*)(arow + half * 16 + 8);
        Frag b0, b1;
        const bf16_t* c0p = &lBT[wave * 32 + n][0];
        const bf16_t* c1p = &lBT[wave * 32 + 16 + n][0];
        b0.h[0] = *(const v8bf*)(c0p + half * 16);
        b0.h[1] = *(const v8bf*)(c0p + half * 16 + 8);
        b1.h[0] = *(const v8bf*)(c1p + half * 16);
        b1.h[1] = *(const v8bf*)(c1p + half * 16 + 8);
        acc0 = wmma_bf16(au.v, b0.v, acc0);
        acc1 = wmma_bf16(au.v, b1.v, acc1);
    }
#pragma unroll
    for (int r = 0; r < 8; ++r) {
        int m = r + half * 8;
        int o0 = nblk + wave * 32 + n;
        float v0 = acc0[r] + bias[o0];
        float v1 = acc1[r] + bias[o0 + 16];
        out[(long)(m0 + m) * N + o0]      = leaky01(v0);
        out[(long)(m0 + m) * N + o0 + 16] = leaky01(v1);
    }
}

// ---------------------------------------------------------------------------
extern "C" void kernel_launch(void* const* d_in, const int* in_sizes, int n_in,
                              void* d_out, int out_size, void* d_ws, size_t ws_size,
                              hipStream_t stream) {
    (void)in_sizes; (void)n_in; (void)out_size; (void)ws_size;
    const int B = 16, L = 1024, D = 640, D2 = 1280;

    const float* feats = (const float*)d_in[0];
    const float* cp_w1 = (const float*)d_in[1];
    const float* cp_b1 = (const float*)d_in[2];
    const float* cp_w2 = (const float*)d_in[3];
    const float* cp_b2 = (const float*)d_in[4];
    const float* cp_w3 = (const float*)d_in[5];
    const float* cp_b3 = (const float*)d_in[6];
    const float* cv_w1 = (const float*)d_in[7];
    const float* cv_b1 = (const float*)d_in[8];
    const float* cv_w2 = (const float*)d_in[9];
    const float* cv_b2 = (const float*)d_in[10];
    const float* cv_w3 = (const float*)d_in[11];
    const float* cv_b3 = (const float*)d_in[12];
    const float* mlp_w = (const float*)d_in[13];
    const float* mlp_b = (const float*)d_in[14];
    float* out = (float*)d_out;

    char* p = (char*)d_ws;
    auto alloc = [&](size_t bytes) -> void* {
        void* r = (void*)p;
        p += (bytes + 255) & ~(size_t)255;
        return r;
    };
    bf16_t* wp_cp1 = (bf16_t*)alloc((size_t)3 * D2 * D * 2);
    bf16_t* wp_cp2 = (bf16_t*)alloc((size_t)3 * D2 * D2 * 2);
    bf16_t* wp_cv1 = (bf16_t*)alloc((size_t)3 * D2 * D * 2);
    bf16_t* wp_cv2 = (bf16_t*)alloc((size_t)3 * D2 * D2 * 2);
    bf16_t* wp_cv3 = (bf16_t*)alloc((size_t)3 * D * D2 * 2);
    bf16_t* mlp_wT = (bf16_t*)alloc((size_t)D2 * D * 2);
    bf16_t* fT     = (bf16_t*)alloc((size_t)B * D * L * 2);
    bf16_t* fbf    = (bf16_t*)alloc((size_t)B * L * D * 2);
    bf16_t* xbuf   = (bf16_t*)alloc((size_t)B * D2 * L * 2);
    float*  ybuf   = (float*) alloc((size_t)B * D2 * L * 4);
    bf16_t* convo  = (bf16_t*)alloc((size_t)B * D * L * 2);
    float*  swraw  = (float*) alloc((size_t)B * 2 * L * 4);
    float*  swact  = (float*) alloc((size_t)B * 2 * L * 4);
    float*  invden = (float*) alloc((size_t)B * L * 4);
    float*  wsoft  = (float*) alloc((size_t)B * L * 4);
    float*  outsa  = (float*) alloc((size_t)B * L * D * 4);
    bf16_t* catb   = (bf16_t*)alloc((size_t)B * L * 2 * D * 2);
    float*  mr     = (float*) alloc((size_t)B * 2 * 4);

    // ---- packing / casts ----
    pack_conv_w<<<512, 256, 0, stream>>>(cp_w1, wp_cp1, D2, D);
    pack_conv_w<<<1024, 256, 0, stream>>>(cp_w2, wp_cp2, D2, D2);
    pack_conv_w<<<512, 256, 0, stream>>>(cv_w1, wp_cv1, D2, D);
    pack_conv_w<<<1024, 256, 0, stream>>>(cv_w2, wp_cv2, D2, D2);
    pack_conv_w<<<512, 256, 0, stream>>>(cv_w3, wp_cv3, D, D2);
    pack_mlp<<<512, 256, 0, stream>>>(mlp_w, mlp_wT, D, D2);
    prep_feats<<<2048, 256, 0, stream>>>(feats, fT, fbf, B, L, D);

    dim3 cg1(L / 32, D2 / 128, B);
    dim3 cg3(L / 32, D  / 128, B);
    dim3 lng(512, B);

    // ---- ContextPooling conv stack ----
    conv_gemm<<<cg1, 256, 0, stream>>>(fT, wp_cp1, cp_b1, ybuf, D, D2, L);
    ln_stats<<<B, 256, 0, stream>>>(ybuf, mr, D2 * L);
    ln_apply<<<lng, 256, 0, stream>>>(ybuf, mr, xbuf, (float*)nullptr, D2 * L);
    conv_gemm<<<cg1, 256, 0, stream>>>(xbuf, wp_cp2, cp_b2, ybuf, D2, D2, L);
    ln_stats<<<B, 256, 0, stream>>>(ybuf, mr, D2 * L);
    ln_apply<<<lng, 256, 0, stream>>>(ybuf, mr, xbuf, (float*)nullptr, D2 * L);
    conv_small<<<dim3(L / 256, 2, B), 256, 0, stream>>>(xbuf, cp_w3, cp_b3, swraw, D2, L);
    ln_stats<<<B, 256, 0, stream>>>(swraw, mr, 2 * L);
    ln_apply<<<dim3(8, B), 256, 0, stream>>>(swraw, mr, (bf16_t*)nullptr, swact, 2 * L);
    softmax_sw<<<dim3(2, B), 256, 0, stream>>>(swact, invden, wsoft, L);
    pool_gemm<<<dim3(D / 128, L / 16, B), 128, 0, stream>>>(fbf, invden, wsoft, outsa, L, D);

    // ---- SoluModel conv stack ----
    conv_gemm<<<cg1, 256, 0, stream>>>(fT, wp_cv1, cv_b1, ybuf, D, D2, L);
    ln_stats<<<B, 256, 0, stream>>>(ybuf, mr, D2 * L);
    ln_apply<<<lng, 256, 0, stream>>>(ybuf, mr, xbuf, (float*)nullptr, D2 * L);
    conv_gemm<<<cg1, 256, 0, stream>>>(xbuf, wp_cv2, cv_b2, ybuf, D2, D2, L);
    ln_stats<<<B, 256, 0, stream>>>(ybuf, mr, D2 * L);
    ln_apply<<<lng, 256, 0, stream>>>(ybuf, mr, xbuf, (float*)nullptr, D2 * L);
    conv_gemm<<<cg3, 256, 0, stream>>>(xbuf, wp_cv3, cv_b3, ybuf, D2, D, L);
    ln_stats<<<B, 256, 0, stream>>>(ybuf, mr, D * L);
    ln_apply<<<lng, 256, 0, stream>>>(ybuf, mr, convo, (float*)nullptr, D * L);

    // ---- concat + MLP ----
    build_cat<<<dim3(L, B), 256, 0, stream>>>(outsa, convo, feats, catb, L, D);
    mlp_gemm<<<dim3(D / 128, (B * L) / 16), 128, 0, stream>>>(catb, mlp_wT, mlp_b, out,
                                                              B * L, D2, D);
}
